// PointUpsampling_46600395161947
// MI455X (gfx1250) — matrix operand order repaired and checked by reading, not compiled, not run on hardware
//
#include <hip/hip_runtime.h>
#include <math.h>

typedef __bf16 bf16;
typedef __attribute__((ext_vector_type(16))) __bf16 v16bf;
typedef __attribute__((ext_vector_type(2)))  __bf16 bf16x2;
typedef __attribute__((ext_vector_type(4)))  __bf16 bf16x4;
typedef __attribute__((ext_vector_type(8)))  float  v8f;

#define BATCH 8
#define SPTS  2048
#define NPTS  8192
#define FEAT  256
#define CH    128
#define DIN   384
#define DH    512
#define MROWS 65536          // BATCH*NPTS
#define EPSF  1.1920929e-07f
#define BNEPS 1e-5f

// ---------------------------------------------------------------- zero stats
__global__ __launch_bounds__(256) void zero_kernel(float* p, int n) {
    int i = blockIdx.x * 256 + threadIdx.x;
    if (i < n) p[i] = 0.0f;
}

// -------------------------------------------------- convert weights to bf16
__global__ __launch_bounds__(256) void convw_kernel(const float* __restrict__ W1,
                                                    const float* __restrict__ W2,
                                                    bf16* __restrict__ W1b,
                                                    bf16* __restrict__ W2b) {
    const int n1 = DIN * DH;           // 196608
    const int n2 = DH * DH;            // 262144
    int i = blockIdx.x * 256 + threadIdx.x;
    if (i < n1) W1b[i] = (bf16)W1[i];
    else if (i < n1 + n2) W2b[i - n1] = (bf16)W2[i - n1];
}

// --------------------- top-3 NN + inverse-distance interp + concat -> A1 bf16
// Phase 1: one thread per point -> top-3 idx + normalized weights into LDS.
// Phase 2: lanes stride the feature dim -> coalesced reads & bf16x2 writes.
__global__ __launch_bounds__(256) void interp_kernel(const float* __restrict__ sxyz,
                                                     const float* __restrict__ sfeat,
                                                     const float* __restrict__ xyz,
                                                     const float* __restrict__ pfeat,
                                                     bf16* __restrict__ A1) {
    __shared__ float sx[SPTS], sy[SPTS], sz[SPTS];
    __shared__ int   sidx[256][3];
    __shared__ float swgt[256][3];

    const int tid   = threadIdx.x;
    const int batch = blockIdx.x >> 5;                 // 32 blocks per batch
    const int pbase = (blockIdx.x & 31) << 8;          // first point of block

    const float* sb = sxyz + (size_t)batch * SPTS * 3;
    for (int i = tid; i < SPTS; i += 256) {
        sx[i] = sb[i * 3 + 0];
        sy[i] = sb[i * 3 + 1];
        sz[i] = sb[i * 3 + 2];
    }
    __syncthreads();

    // ---- phase 1: per-point top-3 search
    {
        const int p = batch * NPTS + pbase + tid;
        const float px = xyz[(size_t)p * 3 + 0];
        const float py = xyz[(size_t)p * 3 + 1];
        const float pz = xyz[(size_t)p * 3 + 2];

        float d0 = 3.0e38f, d1 = 3.0e38f, d2 = 3.0e38f;
        int   i0 = 0, i1 = 0, i2 = 0;
        for (int s = 0; s < SPTS; ++s) {
            float dx = px - sx[s], dy = py - sy[s], dz = pz - sz[s];
            float d  = dx * dx + dy * dy + dz * dz;
            if (d < d2) {
                if (d < d1) {
                    if (d < d0) { d2 = d1; i2 = i1; d1 = d0; i1 = i0; d0 = d; i0 = s; }
                    else        { d2 = d1; i2 = i1; d1 = d;  i1 = s; }
                } else          { d2 = d;  i2 = s; }
            }
        }
        float w0 = 1.0f / (fmaxf(d0, 0.0f) + EPSF);
        float w1 = 1.0f / (fmaxf(d1, 0.0f) + EPSF);
        float w2 = 1.0f / (fmaxf(d2, 0.0f) + EPSF);
        float wi = 1.0f / (w0 + w1 + w2);
        sidx[tid][0] = i0; sidx[tid][1] = i1; sidx[tid][2] = i2;
        swgt[tid][0] = w0 * wi; swgt[tid][1] = w1 * wi; swgt[tid][2] = w2 * wi;
    }
    __syncthreads();

    // ---- phase 2: cooperative, coalesced row assembly
    // 256 points x 192 bf16-pairs per row = 49152 work units.
    const size_t fb = (size_t)batch * SPTS;
    for (int u = tid; u < 256 * (DIN / 2); u += 256) {
        const int pt   = u / (DIN / 2);
        const int col2 = u - pt * (DIN / 2);           // bf16 pair index
        const int p    = batch * NPTS + pbase + pt;
        bf16x2 outv;
        if (col2 < CH / 2) {
            const float2 v = ((const float2*)(pfeat + (size_t)p * CH))[col2];
            outv[0] = (bf16)v.x; outv[1] = (bf16)v.y;
        } else {
            const int f2 = col2 - CH / 2;              // float2 index into FEAT
            const float w0 = swgt[pt][0], w1 = swgt[pt][1], w2 = swgt[pt][2];
            const float2 a = ((const float2*)(sfeat + (fb + sidx[pt][0]) * FEAT))[f2];
            const float2 b = ((const float2*)(sfeat + (fb + sidx[pt][1]) * FEAT))[f2];
            const float2 c = ((const float2*)(sfeat + (fb + sidx[pt][2]) * FEAT))[f2];
            outv[0] = (bf16)(w0 * a.x + w1 * b.x + w2 * c.x);
            outv[1] = (bf16)(w0 * a.y + w1 * b.y + w2 * c.y);
        }
        *(bf16x2*)&A1[(size_t)p * DIN + 2 * col2] = outv;
    }
}

// ---------------------------------------------------------- bf16 WMMA GEMM
// Y[M x 512] = A[M x K] * W[K x 512].  Block tile 128(M) x 32(N), BK=32.
// 8 waves/block: wave w owns rows [w*16, w*16+16), two 16x16 N-tiles.
__global__ __launch_bounds__(256) void gemm_bf16_wmma(const bf16* __restrict__ A,
                                                      const bf16* __restrict__ W,
                                                      float* __restrict__ Y,
                                                      int K) {
    __shared__ __align__(16) bf16 sA[128 * 32];   // [row][k]
    __shared__ __align__(16) bf16 sB[32 * 32];    // [n][k]  (transposed)

    const int tid  = threadIdx.x;
    const int lane = tid & 31;
    const int wave = tid >> 5;
    const int m0   = blockIdx.y * 128;
    const int n0   = blockIdx.x * 32;

    v8f c0 = {}; v8f c1 = {};

    const int rowA  = (wave << 4) + (lane & 15);
    const int kbase = (lane < 16) ? 0 : 8;    // A: lanes 16-31 start at K=8
    const int khalf = (lane < 16) ? 0 : 16;   // unused; B reads use kbase (lanes 16-31: K=8..15,24..31)
    const int colB  = lane & 15;

    for (int kb = 0; kb < K; kb += 32) {
        // A tile 128x32 bf16: one uint4 (8 bf16 at k=h..h+7, h=0 or 16) per thread
        {
            int r = tid >> 1, h = (tid & 1) * 16;
            *(uint4*)&sA[r * 32 + h] =
                *(const uint4*)&A[(size_t)(m0 + r) * K + kb + h];
        }
        // B tile transposed: sB[n][k] = W[kb+k][n0+n]
        {
            int n = tid >> 3, kq = (tid & 7) * 4;
#pragma unroll
            for (int i = 0; i < 4; ++i)
                sB[n * 32 + kq + i] = W[(size_t)(kb + kq + i) * DH + n0 + n];
        }
        __syncthreads();

        // A fragment (16x32 bf16, ISA lane layout)
        v16bf a;
#pragma unroll
        for (int j = 0; j < 8; ++j) {
            int k = kbase + ((j < 4) ? 2 * j : 8 + 2 * j);
            bf16x2 pr = *(const bf16x2*)&sA[rowA * 32 + k];
            a[2 * j] = pr[0]; a[2 * j + 1] = pr[1];
        }
        // B fragments for the two 16-wide N tiles
        v16bf b0, b1;
#pragma unroll
        for (int j = 0; j < 8; ++j) {
            int k = kbase + ((j < 4) ? 2 * j : 8 + 2 * j);
            bf16x2 p0 = *(const bf16x2*)&sB[colB * 32 + k];
            bf16x2 p1 = *(const bf16x2*)&sB[(16 + colB) * 32 + k];
            b0[2 * j] = p0[0]; b0[2 * j + 1] = p0[1];
            b1[2 * j] = p1[0]; b1[2 * j + 1] = p1[1];
        }

        c0 = __builtin_amdgcn_wmma_f32_16x16x32_bf16(false, a, false, b0,
                                                     (short)0, c0, false, false);
        c1 = __builtin_amdgcn_wmma_f32_16x16x32_bf16(false, a, false, b1,
                                                     (short)0, c1, false, false);
        __syncthreads();
    }

    // D layout: vgpr r -> lanes 0-15: (M=r, N=lane); lanes 16-31: (M=r+8, N=lane-16)
    const int mrow = m0 + (wave << 4) + ((lane < 16) ? 0 : 8);
    const int ncol = n0 + (lane & 15);
#pragma unroll
    for (int r = 0; r < 8; ++r) {
        Y[(size_t)(mrow + r) * DH + ncol]      = c0[r];
        Y[(size_t)(mrow + r) * DH + ncol + 16] = c1[r];
    }
}

// ------------------------------------------- per-channel sum / sumsq (BN stats)
__global__ __launch_bounds__(256) void stats_kernel(const float* __restrict__ Y,
                                                    float* __restrict__ gsum,
                                                    float* __restrict__ gsq) {
    __shared__ float ss[DH], sq[DH];
    const int tid = threadIdx.x;
    for (int i = tid; i < DH; i += 256) { ss[i] = 0.0f; sq[i] = 0.0f; }
    __syncthreads();

    const size_t total4  = (size_t)MROWS * DH / 4;
    const size_t stride  = (size_t)gridDim.x * 256;
    const float4* Y4 = (const float4*)Y;
    for (size_t i = (size_t)blockIdx.x * 256 + tid; i < total4; i += stride) {
        float4 v = Y4[i];
        int ch = (int)((i * 4) & (DH - 1));
        atomicAdd(&ss[ch + 0], v.x); atomicAdd(&sq[ch + 0], v.x * v.x);
        atomicAdd(&ss[ch + 1], v.y); atomicAdd(&sq[ch + 1], v.y * v.y);
        atomicAdd(&ss[ch + 2], v.z); atomicAdd(&sq[ch + 2], v.z * v.z);
        atomicAdd(&ss[ch + 3], v.w); atomicAdd(&sq[ch + 3], v.w * v.w);
    }
    __syncthreads();
    for (int i = tid; i < DH; i += 256) {
        atomicAdd(&gsum[i], ss[i]);
        atomicAdd(&gsq[i],  sq[i]);
    }
}

__device__ __forceinline__ float bn_gelu_elem(float y, float s, float s2,
                                              float g, float b) {
    const float inv = 1.0f / (float)MROWS;
    float m = s * inv;
    float v = s2 * inv - m * m;
    float xn = (y - m) * rsqrtf(v + BNEPS);
    float x = xn * g + b;
    return 0.5f * x * (1.0f + erff(x * 0.7071067811865475f));
}

// ------------------------------------------ BN+GELU -> bf16 activations (A2)
__global__ __launch_bounds__(256) void bn_gelu_bf16_kernel(const float* __restrict__ Y,
                                                           const float* __restrict__ gsum,
                                                           const float* __restrict__ gsq,
                                                           const float* __restrict__ gamma,
                                                           const float* __restrict__ beta,
                                                           bf16* __restrict__ out) {
    const size_t total4 = (size_t)MROWS * DH / 4;
    const size_t stride = (size_t)gridDim.x * 256;
    const float4* Y4 = (const float4*)Y;
    for (size_t i = (size_t)blockIdx.x * 256 + threadIdx.x; i < total4; i += stride) {
        float4 v = Y4[i];
        int ch = (int)((i * 4) & (DH - 1));
        bf16x4 o;
        o[0] = (bf16)bn_gelu_elem(v.x, gsum[ch+0], gsq[ch+0], gamma[ch+0], beta[ch+0]);
        o[1] = (bf16)bn_gelu_elem(v.y, gsum[ch+1], gsq[ch+1], gamma[ch+1], beta[ch+1]);
        o[2] = (bf16)bn_gelu_elem(v.z, gsum[ch+2], gsq[ch+2], gamma[ch+2], beta[ch+2]);
        o[3] = (bf16)bn_gelu_elem(v.w, gsum[ch+3], gsq[ch+3], gamma[ch+3], beta[ch+3]);
        *(bf16x4*)&out[i * 4] = o;
    }
}

// ------------------------------------------------- BN+GELU in-place, f32 out
__global__ __launch_bounds__(256) void bn_gelu_f32_kernel(float* __restrict__ Y,
                                                          const float* __restrict__ gsum,
                                                          const float* __restrict__ gsq,
                                                          const float* __restrict__ gamma,
                                                          const float* __restrict__ beta) {
    const size_t total4 = (size_t)MROWS * DH / 4;
    const size_t stride = (size_t)gridDim.x * 256;
    float4* Y4 = (float4*)Y;
    for (size_t i = (size_t)blockIdx.x * 256 + threadIdx.x; i < total4; i += stride) {
        float4 v = Y4[i];
        int ch = (int)((i * 4) & (DH - 1));
        v.x = bn_gelu_elem(v.x, gsum[ch+0], gsq[ch+0], gamma[ch+0], beta[ch+0]);
        v.y = bn_gelu_elem(v.y, gsum[ch+1], gsq[ch+1], gamma[ch+1], beta[ch+1]);
        v.z = bn_gelu_elem(v.z, gsum[ch+2], gsq[ch+2], gamma[ch+2], beta[ch+2]);
        v.w = bn_gelu_elem(v.w, gsum[ch+3], gsq[ch+3], gamma[ch+3], beta[ch+3]);
        Y4[i] = v;
    }
}

extern "C" void kernel_launch(void* const* d_in, const int* in_sizes, int n_in,
                              void* d_out, int out_size, void* d_ws, size_t ws_size,
                              hipStream_t stream) {
    const float* sxyz  = (const float*)d_in[0];
    const float* sfeat = (const float*)d_in[1];
    const float* xyz   = (const float*)d_in[2];
    const float* pfeat = (const float*)d_in[3];
    const float* W1    = (const float*)d_in[4];
    const float* g1    = (const float*)d_in[5];
    const float* b1    = (const float*)d_in[6];
    const float* W2    = (const float*)d_in[7];
    const float* g2    = (const float*)d_in[8];
    const float* b2    = (const float*)d_in[9];
    float* out = (float*)d_out;

    // workspace layout (bytes)
    char* ws = (char*)d_ws;
    bf16*  A1    = (bf16*)(ws + 0);                       //  50,331,648
    bf16*  A2    = (bf16*)(ws + 50331648ull);             //  67,108,864
    float* Y1    = (float*)(ws + 117440512ull);           // 134,217,728
    bf16*  W1b   = (bf16*)(ws + 251658240ull);            //     393,216
    bf16*  W2b   = (bf16*)(ws + 252051456ull);            //     524,288
    float* stats = (float*)(ws + 252575744ull);           // 4*512 floats

    zero_kernel<<<8, 256, 0, stream>>>(stats, 4 * DH);

    const int nw = DIN * DH + DH * DH;
    convw_kernel<<<(nw + 255) / 256, 256, 0, stream>>>(W1, W2, W1b, W2b);

    interp_kernel<<<256, 256, 0, stream>>>(sxyz, sfeat, xyz, pfeat, A1);

    dim3 ggrid(DH / 32, MROWS / 128);   // 16 x 512
    gemm_bf16_wmma<<<ggrid, 256, 0, stream>>>(A1, W1b, Y1, DIN);

    stats_kernel<<<512, 256, 0, stream>>>(Y1, stats, stats + DH);
    bn_gelu_bf16_kernel<<<2048, 256, 0, stream>>>(Y1, stats, stats + DH, g1, b1, A2);

    gemm_bf16_wmma<<<ggrid, 256, 0, stream>>>(A2, W2b, out, DH);

    stats_kernel<<<512, 256, 0, stream>>>(out, stats + 2 * DH, stats + 3 * DH);
    bn_gelu_f32_kernel<<<2048, 256, 0, stream>>>(out, stats + 2 * DH, stats + 3 * DH, g2, b2);
}
